// FitLoss_57475252355292
// MI455X (gfx1250) — matrix operand order repaired and checked by reading, not compiled, not run on hardware
//
#include <hip/hip_runtime.h>
#include <hip/hip_bf16.h>
#include <math.h>

// ---------------------------------------------------------------------------
// FitLoss on MI455X (gfx1250): fused WMMA-GEMM + weighted exp-sum + log dot.
//   S[m]   = sum_n x_w[n] * exp( min(0, (2*x.y - |x|^2 - |y|^2)) / 128 )
//   result = -sum_m log(S[m]) * y_w[m]
// Single pass (terms bounded in (0,1] -> no LSE max pass needed).
// GEMM in split-bf16 (hi+lo) with f32 accumulation: 3 wmma per tile.
// x chunks streamed global->LDS with the Tensor Data Mover (double-buffered,
// TENSORcnt-tracked), with TDM LDS padding giving a conflict-free 132-dword
// row stride. hi/lo split of x happens at fragment-load time in registers.
// ---------------------------------------------------------------------------

typedef __attribute__((ext_vector_type(16))) __bf16 v16bf;
typedef __attribute__((ext_vector_type(8)))  __bf16 v8bf;
typedef __attribute__((ext_vector_type(8)))  float  v8f;
typedef __attribute__((ext_vector_type(4)))  unsigned int v4u;
typedef __attribute__((ext_vector_type(8)))  int v8i;
typedef __attribute__((ext_vector_type(4)))  int v4i;

#define NN 8192
#define MM 16384
#define DD 128
#define BN 128          // m-columns per workgroup
#define BMROWS 64       // n-rows per TDM chunk
#define NSPLIT 4        // grid split along n (partials via f32 atomics)
#define LDK (DD + 16)   // bf16 row stride for y tiles (288 B, 16B aligned)
#define XLDW (DD + 4)   // f32 dword row stride for TDM-staged x (528 B rows)

__device__ __forceinline__ float inv2s2() { return 1.0f / 128.0f; } // 1/(2*(8*1)^2)

__device__ __forceinline__ unsigned pack2(__bf16 a, __bf16 b) {
  union { __bf16 h; unsigned short u; } ca, cb;
  ca.h = a; cb.h = b;
  return (unsigned)ca.u | ((unsigned)cb.u << 16);
}

// TDM descriptor + issue: DMA a [BMROWS x DD] f32 tile, row-major, into LDS
// with 4 dwords of padding appended per 128-dword row (bank-conflict-free).
// This toolchain exposes the 6-arg builtin:
//   (uint32x4 g0, int32x8 g1, int32x4 g2, int32x4 g3, int32x8 extra, i32 cpol)
__device__ __forceinline__ void tdm_load_x(const float* gsrc, unsigned lds_off) {
  unsigned long long ga = (unsigned long long)(uintptr_t)gsrc;
  // D# group0: count=1 | lds_addr | global_addr[56:0] | type=2 ("image")
  v4u g0 = { 1u,
             lds_off,
             (unsigned)(ga & 0xffffffffu),
             (unsigned)((ga >> 32) & 0x01ffffffu) | (2u << 30) };
  // D# group1: data_size=4B(2), pad_enable, pad_interval=6 (128 dw),
  // pad_amount=3 (4 dw); tensor 128x64, tile 128x64, dim0_stride=128.
  v8i g1 = { (int)((2u << 16) | (1u << 20) | (6u << 22) | (3u << 25)),
             (int)(128u << 16),   // tensor_dim0 = DD  (bits 79:48)
             (int)(64u  << 16),   // tensor_dim1 = BMROWS (bits 111:80)
             (int)(128u << 16),   // tile_dim0 = DD (bits 127:112)
             (int)64,             // tile_dim1 = BMROWS; tile_dim2 = 0
             (int)128,            // tensor_dim0_stride = DD (low 32)
             0, 0 };
  v4i gz4 = { 0, 0, 0, 0 };         // groups 2/3: dims 2+ unused (2D tensor)
  v8i gz8 = { 0, 0, 0, 0, 0, 0, 0, 0 };
  __builtin_amdgcn_tensor_load_to_lds(g0, g1, gz4, gz4, gz8, 0);
}

// --------------------------- row norms -------------------------------------
__global__ __launch_bounds__(256) void prep_norms(
    const float* __restrict__ x, const float* __restrict__ y,
    float* __restrict__ x2, float* __restrict__ y2) {
  int i = blockIdx.x * 256 + threadIdx.x;
  const float* src;
  float* dst;
  int row;
  if (i < NN)            { src = x; dst = x2; row = i; }
  else if (i < NN + MM)  { src = y; dst = y2; row = i - NN; }
  else return;
  const float4* p = (const float4*)(src + (size_t)row * DD);
  float s = 0.f;
#pragma unroll
  for (int j = 0; j < DD / 4; ++j) {
    float4 v = p[j];
    s += v.x * v.x + v.y * v.y + v.z * v.z + v.w * v.w;
  }
  dst[row] = s;
}

// --------------------- fused GEMM + exp-sum over n -------------------------
__global__ __launch_bounds__(256) void gemm_expsum(
    const float* __restrict__ x, const float* __restrict__ xw,
    const float* __restrict__ y,
    const float* __restrict__ x2g, const float* __restrict__ y2g,
    float* __restrict__ S) {
  __shared__ __bf16 ysh[BN][LDK];                 // y tile hi
  __shared__ __bf16 ysl[BN][LDK];                 // y tile lo
  __shared__ float  xstage[2][BMROWS * XLDW];     // TDM-staged f32 x chunks
  __shared__ float  x2s[BMROWS];
  __shared__ float  xws[BMROWS];

  const int tid  = threadIdx.x;
  const int lane = tid & 31;
  const int wave = tid >> 5;
  const int wn   = wave & 3;   // n sub-tile of the 64-row chunk (16 rows each)
  const int wm   = wave >> 2;  // m half (64 cols each)
  const int lh   = lane >> 4;  // lane half (WMMA K-half selector)
  const int lc   = lane & 15;  // column / row within 16

  const int mblk = blockIdx.x * BN;
  const int nb0  = blockIdx.y * (NN / NSPLIT);
  const int iters = (NN / NSPLIT) / BMROWS;

  // Kick off TDM for chunk 0 immediately (wave 0 only; TDM ignores EXEC,
  // waves 1..7 branch around).
  if (wave == 0) {
    tdm_load_x(x + (size_t)nb0 * DD, (unsigned)(uintptr_t)&xstage[0][0]);
  }

  // Stage y tile [BN x DD] once, split into bf16 hi/lo (packed b64 stores).
#pragma unroll
  for (int j = 0; j < (BN * DD / 4) / 256; ++j) {
    int idx = tid + 256 * j;
    int row = idx >> 5;          // 32 float4 per row
    int k   = (idx & 31) * 4;
    float4 v = *(const float4*)(y + (size_t)(mblk + row) * DD + k);
    float f[4] = {v.x, v.y, v.z, v.w};
    __bf16 h[4], l[4];
#pragma unroll
    for (int q = 0; q < 4; ++q) {
      h[q] = (__bf16)f[q];
      l[q] = (__bf16)(f[q] - (float)h[q]);
    }
    uint2 ph = make_uint2(pack2(h[0], h[1]), pack2(h[2], h[3]));
    uint2 pl = make_uint2(pack2(l[0], l[1]), pack2(l[2], l[3]));
    *(uint2*)&ysh[row][k] = ph;
    *(uint2*)&ysl[row][k] = pl;
  }

  // per-lane column norms (column is fixed per tile for the whole n loop)
  float y2c[4];
#pragma unroll
  for (int t = 0; t < 4; ++t) y2c[t] = y2g[mblk + wm * 64 + 16 * t + lc];

  float colsum[4] = {0.f, 0.f, 0.f, 0.f};
  const int arow = wn * 16 + lc;  // A-frag row (same for both lane halves)

  for (int it = 0; it < iters; ++it) {
    const int buf = it & 1;
    const int nb  = nb0 + it * BMROWS;

    if (tid < BMROWS) {
      x2s[tid] = x2g[nb + tid];
      xws[tid] = xw[nb + tid];
    }
    // Pipeline: issue next chunk's DMA into the other buffer, then wait for
    // the current chunk (TDM completes in order per wave -> wait tensorcnt<=1).
    if (wave == 0) {
      if (it + 1 < iters) {
        tdm_load_x(x + (size_t)(nb + BMROWS) * DD,
                   (unsigned)(uintptr_t)&xstage[buf ^ 1][0]);
        __builtin_amdgcn_s_wait_tensorcnt(1);
      } else {
        __builtin_amdgcn_s_wait_tensorcnt(0);
      }
    }
    __syncthreads();  // publish chunk `it` LDS tile (+ x2s/xws) to all waves

    v8f acc[4] = {};  // 4 m-tiles of 16x16 f32
    const float* xrow = &xstage[buf][arow * XLDW];

#pragma unroll
    for (int kb = 0; kb < DD; kb += 32) {
      // A 16x32 frag: elems 0..7 = K kb+8*lh+0..7, elems 8..15 = +16.
      // Read f32 from TDM-staged LDS, split hi/lo bf16 in registers.
      float fv[16];
      *(float4*)&fv[0]  = *(const float4*)(xrow + kb + 8 * lh);
      *(float4*)&fv[4]  = *(const float4*)(xrow + kb + 8 * lh + 4);
      *(float4*)&fv[8]  = *(const float4*)(xrow + kb + 8 * lh + 16);
      *(float4*)&fv[12] = *(const float4*)(xrow + kb + 8 * lh + 20);
      union { v16bf v; __bf16 e[16]; } Ah, Al;
#pragma unroll
      for (int i = 0; i < 16; ++i) {
        __bf16 h = (__bf16)fv[i];
        Ah.e[i] = h;
        Al.e[i] = (__bf16)(fv[i] - (float)h);
      }
#pragma unroll
      for (int t = 0; t < 4; ++t) {
        const int brow = wm * 64 + 16 * t + lc;  // B column = y row (LDS [m][k])
        union { v16bf v; v8bf h[2]; } Bh, Bl;
        Bh.h[0] = *(const v8bf*)&ysh[brow][kb + 8 * lh];
        Bh.h[1] = *(const v8bf*)&ysh[brow][kb + 8 * lh + 16];
        Bl.h[0] = *(const v8bf*)&ysl[brow][kb + 8 * lh];
        Bl.h[1] = *(const v8bf*)&ysl[brow][kb + 8 * lh + 16];
        acc[t] = __builtin_amdgcn_wmma_f32_16x16x32_bf16(
            false, Ah.v, false, Bh.v, (short)0, acc[t], false, false);
        acc[t] = __builtin_amdgcn_wmma_f32_16x16x32_bf16(
            false, Al.v, false, Bh.v, (short)0, acc[t], false, false);
        acc[t] = __builtin_amdgcn_wmma_f32_16x16x32_bf16(
            false, Ah.v, false, Bl.v, (short)0, acc[t], false, false);
      }
    }

    // Epilogue: C VGPR r holds (row = wn*16 + r + 8*lh, col = lc) per lane.
    float x2v[8], xwv[8];
#pragma unroll
    for (int r = 0; r < 8; ++r) {
      int nl = wn * 16 + 8 * lh + r;
      x2v[r] = x2s[nl];
      xwv[r] = xws[nl];
    }
#pragma unroll
    for (int t = 0; t < 4; ++t) {
      float p = 0.f;
#pragma unroll
      for (int r = 0; r < 8; ++r) {
        float arg = fminf(0.f, (2.f * acc[t][r] - x2v[r] - y2c[t]) * inv2s2());
        p += xwv[r] * __expf(arg);
      }
      colsum[t] += p;
    }
    __syncthreads();  // all reads of xstage[buf]/x2s done before reuse
  }

  // Merge lane halves (rows 0-7 with rows 8-15 of each wave block) and emit.
#pragma unroll
  for (int t = 0; t < 4; ++t) {
    float s = colsum[t];
    s += __shfl_xor(s, 16, 32);
    if (lane < 16) atomicAdd(&S[mblk + wm * 64 + 16 * t + lane], s);
  }
}

// --------------------------- finalize --------------------------------------
__global__ __launch_bounds__(256) void finalize(
    const float* __restrict__ S, const float* __restrict__ yw,
    float* __restrict__ out) {
  int m = blockIdx.x * 256 + threadIdx.x;
  float v = 0.f;
  if (m < MM) v = -__logf(S[m]) * yw[m];
#pragma unroll
  for (int off = 16; off > 0; off >>= 1) v += __shfl_down(v, off, 32);
  __shared__ float ws[8];
  int lane = threadIdx.x & 31, wv = threadIdx.x >> 5;
  if (lane == 0) ws[wv] = v;
  __syncthreads();
  if (wv == 0) {
    v = (lane < 8) ? ws[lane] : 0.f;
#pragma unroll
    for (int off = 4; off > 0; off >>= 1) v += __shfl_down(v, off, 32);
    if (lane == 0) atomicAdd(out, v);
  }
}

// --------------------------- launcher --------------------------------------
extern "C" void kernel_launch(void* const* d_in, const int* in_sizes, int n_in,
                              void* d_out, int out_size, void* d_ws, size_t ws_size,
                              hipStream_t stream) {
  (void)in_sizes; (void)n_in; (void)out_size; (void)ws_size;
  const float* x  = (const float*)d_in[0];
  const float* xw = (const float*)d_in[1];
  const float* y  = (const float*)d_in[2];
  const float* yw = (const float*)d_in[3];

  float* S   = (float*)d_ws;   // MM floats
  float* x2  = S + MM;         // NN floats
  float* y2  = x2 + NN;        // MM floats
  float* out = (float*)d_out;

  (void)hipMemsetAsync(S, 0, MM * sizeof(float), stream);
  (void)hipMemsetAsync(out, 0, sizeof(float), stream);

  prep_norms<<<(NN + MM + 255) / 256, 256, 0, stream>>>(x, y, x2, y2);

  dim3 grid(MM / BN, NSPLIT);
  gemm_expsum<<<grid, 256, 0, stream>>>(x, xw, y, x2, y2, S);

  finalize<<<(MM + 255) / 256, 256, 0, stream>>>(S, yw, out);
}
